// GlimpseAttention_29300266893482
// MI455X (gfx1250) — matrix operand order repaired and checked by reading, not compiled, not run on hardware
//
#include <hip/hip_runtime.h>
#include <stdint.h>

#define B_SZ 32
#define L_SZ 2048
#define H_SZ 512

typedef __attribute__((ext_vector_type(16))) __bf16 v16bf;
typedef __attribute__((ext_vector_type(8)))  __bf16 v8bf;
typedef __attribute__((ext_vector_type(4)))  __bf16 v4bf;
typedef __attribute__((ext_vector_type(8)))  float  v8f;

struct bfpair { __bf16 hi; __bf16 lo; };

__device__ __forceinline__ bfpair split_bf16(float x) {
    bfpair p;
    p.hi = (__bf16)x;
    p.lo = (__bf16)(x - (float)p.hi);
    return p;
}

// ---------------------------------------------------------------------------
// Kernel 0: one-shot hi/lo bf16 split of W_ref (runs once per launch; W_ref
// is reused by all 4096 score blocks, so the split must not be in the K-loop)
// ---------------------------------------------------------------------------
__global__ __launch_bounds__(256)
void wsplit_kernel(const float* __restrict__ W,
                   __bf16* __restrict__ Whi,
                   __bf16* __restrict__ Wlo) {
    const int idx = blockIdx.x * 256 + threadIdx.x;       // float4 index
    const float4 w = reinterpret_cast<const float4*>(W)[idx];
    v4bf h, l;
    bfpair p0 = split_bf16(w.x); h[0] = p0.hi; l[0] = p0.lo;
    bfpair p1 = split_bf16(w.y); h[1] = p1.hi; l[1] = p1.lo;
    bfpair p2 = split_bf16(w.z); h[2] = p2.hi; l[2] = p2.lo;
    bfpair p3 = split_bf16(w.w); h[3] = p3.hi; l[3] = p3.lo;
    reinterpret_cast<v4bf*>(Whi)[idx] = h;
    reinterpret_cast<v4bf*>(Wlo)[idx] = l;
}

// ---------------------------------------------------------------------------
// Kernel 1: encoded_q[b,o] = sum_h query[b,h] * W_q[o,h]   (32x512, tiny)
// ---------------------------------------------------------------------------
__global__ __launch_bounds__(256)
void encq_kernel(const float* __restrict__ query,
                 const float* __restrict__ W_q,
                 float* __restrict__ eq) {
    const int idx = blockIdx.x * 256 + threadIdx.x;       // b*512 + o
    if (idx >= B_SZ * H_SZ) return;
    const int b = idx >> 9;
    const int o = idx & (H_SZ - 1);
    const float4* q = reinterpret_cast<const float4*>(query + (size_t)b * H_SZ);
    const float4* w = reinterpret_cast<const float4*>(W_q + (size_t)o * H_SZ);
    float s = 0.0f;
    #pragma unroll 4
    for (int k = 0; k < H_SZ / 4; ++k) {
        float4 a = q[k], c = w[k];
        s += a.x * c.x + a.y * c.y + a.z * c.z + a.w * c.w;
    }
    eq[idx] = s;
}

// ---------------------------------------------------------------------------
// Kernel 2: scores[b,l] = (1/sqrt(H)) * sum_o v[o]*tanh( (ref@W_ref^T)[b,l,o]
//                                                        + eq[b,o] ), masked.
// GEMM via bf16 hi/lo split: Ahi*Blo + Alo*Bhi + Ahi*Bhi (~fp32 accurate).
// A tile split ONCE into LDS bf16 planes; B fragments read pre-split from
// global. K-loop body = ds_load_b128 / global b128 loads + 12 WMMAs.
// Block = 8 wave32; each wave owns 4 of 32 N-tiles.
// ---------------------------------------------------------------------------
__global__ __launch_bounds__(256)
void scores_kernel(const float* __restrict__ ref,
                   const __bf16* __restrict__ Whi,
                   const __bf16* __restrict__ Wlo,
                   const float* __restrict__ v,
                   const float* __restrict__ eq,
                   const unsigned char* __restrict__ mask,
                   float* __restrict__ scores_out) {
    const int ltile = blockIdx.x;            // 0..127
    const int b     = blockIdx.y;            // 0..31
    const int tid   = threadIdx.x;
    const int wave  = tid >> 5;              // 0..7
    const int lane  = tid & 31;
    const int nl    = lane & 15;
    const int half  = lane >> 4;             // 0 or 1

    __shared__ __align__(16) __bf16 lds_hi[16 * H_SZ];   // 16 KB
    __shared__ __align__(16) __bf16 lds_lo[16 * H_SZ];   // 16 KB
    __shared__ float lds_part[8][16];

    // cooperative load + one-time hi/lo split of the 16x512 ref tile
    {
        const float4* src = reinterpret_cast<const float4*>(
            ref + ((size_t)b * L_SZ + (size_t)ltile * 16) * H_SZ);
        v4bf* dhi = reinterpret_cast<v4bf*>(lds_hi);
        v4bf* dlo = reinterpret_cast<v4bf*>(lds_lo);
        #pragma unroll
        for (int i = 0; i < 8; ++i) {
            float4 t = src[tid + i * 256];
            v4bf h, l;
            bfpair p0 = split_bf16(t.x); h[0] = p0.hi; l[0] = p0.lo;
            bfpair p1 = split_bf16(t.y); h[1] = p1.hi; l[1] = p1.lo;
            bfpair p2 = split_bf16(t.z); h[2] = p2.hi; l[2] = p2.lo;
            bfpair p3 = split_bf16(t.w); h[3] = p3.hi; l[3] = p3.lo;
            dhi[tid + i * 256] = h;
            dlo[tid + i * 256] = l;
        }
    }
    __syncthreads();

    v8f acc[4];
    #pragma unroll
    for (int nt = 0; nt < 4; ++nt)
        #pragma unroll
        for (int j = 0; j < 8; ++j) acc[nt][j] = 0.0f;

    const int m = nl;   // A-matrix row owned by this lane

    for (int kc = 0; kc < 16; ++kc) {        // K = 512 in chunks of 32
        const int k0 = kc * 32;

        // ---- A fragments: ISA 16-bit A 16x32 layout ----
        // lanes 0-15 : elems 0-7 -> K=k0+0..7,  elems 8-15 -> K=k0+16..23
        // lanes 16-31: elems 0-7 -> K=k0+8..15, elems 8-15 -> K=k0+24..31
        const int abase = m * H_SZ + k0 + half * 8;
        v8bf ah0 = *reinterpret_cast<const v8bf*>(&lds_hi[abase]);
        v8bf ah1 = *reinterpret_cast<const v8bf*>(&lds_hi[abase + 16]);
        v8bf al0 = *reinterpret_cast<const v8bf*>(&lds_lo[abase]);
        v8bf al1 = *reinterpret_cast<const v8bf*>(&lds_lo[abase + 16]);
        v16bf a_hi = __builtin_shufflevector(ah0, ah1,
            0,1,2,3,4,5,6,7,8,9,10,11,12,13,14,15);
        v16bf a_lo = __builtin_shufflevector(al0, al1,
            0,1,2,3,4,5,6,7,8,9,10,11,12,13,14,15);

        #pragma unroll
        for (int nt = 0; nt < 4; ++nt) {
            // ---- B fragments: B[k][n] = W_ref[n][k]; ISA B 32x16 layout ----
            // lanes 0-15: K=k0+0..7,16..23; lanes 16-31: K=k0+8..15,24..31 (N=lane)
            const int n = ((wave * 4 + nt) * 16) + nl;
            const size_t wbase = (size_t)n * H_SZ + k0 + half * 8;
            const v8bf* ph = reinterpret_cast<const v8bf*>(Whi + wbase);
            const v8bf* pl = reinterpret_cast<const v8bf*>(Wlo + wbase);
            v16bf b_hi = __builtin_shufflevector(ph[0], ph[2],
                0,1,2,3,4,5,6,7,8,9,10,11,12,13,14,15);
            v16bf b_lo = __builtin_shufflevector(pl[0], pl[2],
                0,1,2,3,4,5,6,7,8,9,10,11,12,13,14,15);

            // emulated-fp32 product: Ahi*Blo + Alo*Bhi + Ahi*Bhi
            acc[nt] = __builtin_amdgcn_wmma_f32_16x16x32_bf16(
                false, a_hi, false, b_lo, (short)0, acc[nt], false, false);
            acc[nt] = __builtin_amdgcn_wmma_f32_16x16x32_bf16(
                false, a_lo, false, b_hi, (short)0, acc[nt], false, false);
            acc[nt] = __builtin_amdgcn_wmma_f32_16x16x32_bf16(
                false, a_hi, false, b_hi, (short)0, acc[nt], false, false);
        }
    }

    // ---- fused epilogue: sacc[m] += v[n] * tanh(S[m][n] + eq[b][n]) ----
    // C/D layout: VGPR j, lanes 0-15 -> (M=j, N=lane); lanes 16-31 -> (M=j+8)
    float sacc[8];
    #pragma unroll
    for (int j = 0; j < 8; ++j) sacc[j] = 0.0f;

    #pragma unroll
    for (int nt = 0; nt < 4; ++nt) {
        const int n = ((wave * 4 + nt) * 16) + nl;
        const float eqv = eq[b * H_SZ + n];
        const float vv  = v[n];
        #pragma unroll
        for (int j = 0; j < 8; ++j)
            sacc[j] += vv * tanhf(acc[nt][j] + eqv);
    }

    // reduce over the 16-lane N group (xor masks stay within each half)
    #pragma unroll
    for (int j = 0; j < 8; ++j) {
        #pragma unroll
        for (int off = 1; off < 16; off <<= 1)
            sacc[j] += __shfl_xor(sacc[j], off, 32);
    }
    if (nl == 0) {
        #pragma unroll
        for (int j = 0; j < 8; ++j)
            lds_part[wave][j + 8 * half] = sacc[j];  // lane0: M=j, lane16: M=j+8
    }
    __syncthreads();

    if (tid < 16) {
        float s = 0.0f;
        #pragma unroll
        for (int w = 0; w < 8; ++w) s += lds_part[w][tid];
        s *= 0.04419417382415922f;                   // 1/sqrt(512)
        const int l = ltile * 16 + tid;
        const bool mk = mask[(size_t)b * L_SZ + l] != 0;
        scores_out[(size_t)b * L_SZ + l] = mk ? -1e9f : s;
    }
}

// ---------------------------------------------------------------------------
// Kernel 3: softmax over L per batch (deterministic LDS tree reductions)
// ---------------------------------------------------------------------------
__global__ __launch_bounds__(256)
void softmax_kernel(const float* __restrict__ scores,
                    float* __restrict__ attn) {
    const int b = blockIdx.x, tid = threadIdx.x;
    __shared__ float red[256];

    float lm = -3.4e38f;
    for (int l = tid; l < L_SZ; l += 256)
        lm = fmaxf(lm, scores[(size_t)b * L_SZ + l]);
    red[tid] = lm; __syncthreads();
    for (int s = 128; s > 0; s >>= 1) {
        if (tid < s) red[tid] = fmaxf(red[tid], red[tid + s]);
        __syncthreads();
    }
    const float mx = red[0];
    __syncthreads();

    float ls = 0.0f;
    for (int l = tid; l < L_SZ; l += 256)
        ls += expf(scores[(size_t)b * L_SZ + l] - mx);
    red[tid] = ls; __syncthreads();
    for (int s = 128; s > 0; s >>= 1) {
        if (tid < s) red[tid] += red[tid + s];
        __syncthreads();
    }
    const float inv = 1.0f / red[0];

    for (int l = tid; l < L_SZ; l += 256)
        attn[(size_t)b * L_SZ + l] = expf(scores[(size_t)b * L_SZ + l] - mx) * inv;
}

// ---------------------------------------------------------------------------
// Kernel 4: glimpse partials over L chunks (no atomics -> deterministic)
// ---------------------------------------------------------------------------
__global__ __launch_bounds__(256)
void glimpse_partial_kernel(const float* __restrict__ ref,
                            const float* __restrict__ attn,
                            float* __restrict__ gpart) {
    const int b = blockIdx.x, chunk = blockIdx.y, tid = threadIdx.x;
    const int h0 = tid, h1 = tid + 256;
    const int lbase = chunk * (L_SZ / 16);
    float a0 = 0.0f, a1 = 0.0f;
    for (int i = 0; i < L_SZ / 16; ++i) {
        const int l = lbase + i;
        const float a = attn[(size_t)b * L_SZ + l];
        const float* r = ref + ((size_t)b * L_SZ + l) * H_SZ;
        a0 += a * r[h0];
        a1 += a * r[h1];
    }
    float* out = gpart + (size_t)(b * 16 + chunk) * H_SZ;
    out[h0] = a0; out[h1] = a1;
}

// ---------------------------------------------------------------------------
// Kernel 5: reduce 16 partials -> glimpse
// ---------------------------------------------------------------------------
__global__ __launch_bounds__(256)
void glimpse_reduce_kernel(const float* __restrict__ gpart,
                           float* __restrict__ glimpse) {
    const int idx = blockIdx.x * 256 + threadIdx.x;
    if (idx >= B_SZ * H_SZ) return;
    const int b = idx >> 9, h = idx & (H_SZ - 1);
    float s = 0.0f;
    #pragma unroll
    for (int c = 0; c < 16; ++c)
        s += gpart[(size_t)(b * 16 + c) * H_SZ + h];
    glimpse[idx] = s;
}

// ---------------------------------------------------------------------------
extern "C" void kernel_launch(void* const* d_in, const int* in_sizes, int n_in,
                              void* d_out, int out_size, void* d_ws, size_t ws_size,
                              hipStream_t stream) {
    const float* query        = (const float*)d_in[0];
    const float* ref          = (const float*)d_in[1];
    const unsigned char* mask = (const unsigned char*)d_in[2];  // jnp bool: 1 byte
    const float* W_ref        = (const float*)d_in[3];
    const float* W_q          = (const float*)d_in[4];
    const float* v            = (const float*)d_in[5];

    float* out     = (float*)d_out;
    float* glimpse = out;                       // B*H
    float* scores  = out + B_SZ * H_SZ;         // B*L

    // workspace layout (byte offsets kept 16B aligned)
    char* ws = (char*)d_ws;
    float*  eq    = (float*)(ws);                                     // 64 KB
    float*  attn  = (float*)(ws + (size_t)B_SZ * H_SZ * 4);           // 256 KB
    float*  gpart = (float*)(ws + (size_t)(B_SZ * H_SZ + B_SZ * L_SZ) * 4);  // 1 MB
    size_t  off   = (size_t)(B_SZ * H_SZ + B_SZ * L_SZ + B_SZ * 16 * H_SZ) * 4;
    __bf16* Whi   = (__bf16*)(ws + off);                              // 512 KB
    __bf16* Wlo   = (__bf16*)(ws + off + (size_t)H_SZ * H_SZ * 2);    // 512 KB

    wsplit_kernel<<<dim3(H_SZ * H_SZ / (256 * 4)), dim3(256), 0, stream>>>(W_ref, Whi, Wlo);
    encq_kernel<<<dim3((B_SZ * H_SZ + 255) / 256), dim3(256), 0, stream>>>(query, W_q, eq);
    scores_kernel<<<dim3(L_SZ / 16, B_SZ), dim3(256), 0, stream>>>(ref, Whi, Wlo, v, eq, mask, scores);
    softmax_kernel<<<dim3(B_SZ), dim3(256), 0, stream>>>(scores, attn);
    glimpse_partial_kernel<<<dim3(B_SZ, 16), dim3(256), 0, stream>>>(ref, attn, gpart);
    glimpse_reduce_kernel<<<dim3((B_SZ * H_SZ + 255) / 256), dim3(256), 0, stream>>>(gpart, glimpse);
}
